// HighPerformanceDecoderModule_54348516164314
// MI455X (gfx1250) — matrix operand order repaired and checked by reading, not compiled, not run on hardware
//
#include <hip/hip_runtime.h>
#include <hip/hip_bf16.h>
#include <math.h>

typedef __bf16 bf16;
typedef __attribute__((ext_vector_type(16))) __bf16 v16bf;
typedef __attribute__((ext_vector_type(8)))  __bf16 v8bf;
typedef __attribute__((ext_vector_type(4)))  __bf16 v4bf;
typedef __attribute__((ext_vector_type(8)))  float  v8f;

#define NWIN   4096
#define NTOK   64      // tokens per window
#define DIM    256
#define NH     8
#define HD     32
#define SCALE  0.17677669529663689f   // 1/sqrt(32)

// ---- bf16 weight workspace layout (transposed: [N][K]) ----
#define WT1_OFF 0           // [512][256]
#define WT2_OFF 131072      // [768][512]
#define WT3_OFF 524288      // [512][256]
#define WT4_OFF 655360      // [256][512]
#define WT_TOTAL 786432

// ---- LDS layout (bytes) ----
#define H1_PITCH  520                      // 512 + 8 pad (bf16 elems)
#define QK_PITCH  528                      // q: 0..255, k: 264..519
#define P_PITCH   72
#define OUT_PITCH 264
#define P_OFF     0                        // 8*64*72*2   = 73728 (h1/h2 alias at 0: 64*520*2 = 66560)
#define QK_OFF    73728                    // 64*528*2    = 67584
#define VT_OFF    141312                   // 8*32*72*2   = 36864
#define OUT_OFF   178176                   // 64*264*2    = 33792  (xb aliases this region pre-attention)
#define ST_OFF    211968                   // psum(1024)+psq(1024)+mean(256)+rstd(256)
#define SMEM_BYTES 214528

// ================= fragment loaders (wave32 WMMA layouts) =================
// A 16x32 bf16: lane l -> row m0+(l&15); elems i = K base+(l>>4)*16+i (contiguous 16)
__device__ __forceinline__ v16bf a_frag_lds(const bf16* base, int pitch, int m0, int k0, int lane) {
    const bf16* p = base + (m0 + (lane & 15)) * pitch + k0 + ((lane >> 4) << 4);
    union { v16bf v; v8bf h[2]; } u;
    u.h[0] = *(const v8bf*)p;
    u.h[1] = *(const v8bf*)(p + 8);
    return u.v;
}
// B 32x16 bf16: lane l -> col n0+(l&15); elems i = K k0+(l>>4)*16+i (contiguous 16)
__device__ __forceinline__ v16bf b_frag_g(const bf16* wt, int K, int n0, int k0, int lane) {
    const bf16* p = wt + (size_t)(n0 + (lane & 15)) * K + k0 + ((lane >> 4) << 4);
    union { v16bf v; v8bf h[2]; } u;
    u.h[0] = *(const v8bf*)p;
    u.h[1] = *(const v8bf*)(p + 8);
    return u.v;
}
__device__ __forceinline__ v16bf b_frag_lds(const bf16* base, int pitch, int n0, int k0, int lane) {
    const bf16* p = base + (n0 + (lane & 15)) * pitch + k0 + ((lane >> 4) << 4);
    union { v16bf v; v8bf h[2]; } u;
    u.h[0] = *(const v8bf*)p;
    u.h[1] = *(const v8bf*)(p + 8);
    return u.v;
}
__device__ __forceinline__ v8f wmma_bf16(v16bf a, v16bf b, v8f c) {
    return __builtin_amdgcn_wmma_f32_16x16x32_bf16(false, a, false, b, (short)0, c, false, false);
}
#define V8F_ZERO (v8f){0.f,0.f,0.f,0.f,0.f,0.f,0.f,0.f}

// ================= LayerNorm + exact GELU, in place on bf16 LDS buffer =================
__device__ __forceinline__ void ln_gelu_inplace(bf16* buf, int pitch, int ncols,
                                                const float* gamma, const float* beta,
                                                float* psum, float* psq, float* meanb, float* rstdb,
                                                int tid) {
    __syncthreads();
    const int row = tid >> 2, seg = tid & 3, segw = ncols >> 2;
    float s = 0.f, q = 0.f;
    for (int c = seg * segw; c < (seg + 1) * segw; ++c) {
        float v = (float)buf[row * pitch + c];
        s += v; q += v * v;
    }
    psum[tid] = s; psq[tid] = q;
    __syncthreads();
    if (tid < 64) {
        float ts = 0.f, tq = 0.f;
#pragma unroll
        for (int k = 0; k < 4; ++k) { ts += psum[tid * 4 + k]; tq += psq[tid * 4 + k]; }
        float m = ts / (float)ncols;
        float v = tq / (float)ncols - m * m;
        meanb[tid] = m;
        rstdb[tid] = rsqrtf(v + 1e-5f);
    }
    __syncthreads();
    const float m = meanb[row], rs = rstdb[row];
    for (int c = seg * segw; c < (seg + 1) * segw; ++c) {
        float v = ((float)buf[row * pitch + c] - m) * rs * gamma[c] + beta[c];
        float g = 0.5f * v * (1.0f + erff(v * 0.70710678118654752f));
        buf[row * pitch + c] = (bf16)g;
    }
    __syncthreads();
}

// ================= weight prep: fp32 [K][N] -> bf16 [N][K] in workspace =================
__global__ void prep_weights(const float* __restrict__ w1, const float* __restrict__ w2,
                             const float* __restrict__ w3, const float* __restrict__ w4,
                             bf16* __restrict__ ws) {
    int i = blockIdx.x * blockDim.x + threadIdx.x;
    if (i < 512 * 256) { int n = i / 256, k = i % 256; ws[WT1_OFF + i] = (bf16)w1[k * 512 + n]; return; }
    i -= 512 * 256;
    if (i < 768 * 512) { int n = i / 512, k = i % 512; ws[WT2_OFF + i] = (bf16)w2[k * 768 + n]; return; }
    i -= 768 * 512;
    if (i < 512 * 256) { int n = i / 256, k = i % 256; ws[WT3_OFF + i] = (bf16)w3[k * 512 + n]; return; }
    i -= 512 * 256;
    if (i < 256 * 512) { int n = i / 512, k = i % 512; ws[WT4_OFF + i] = (bf16)w4[k * 256 + n]; }
}

// ================= fused window-attention block =================
__global__ void __launch_bounds__(256)
fused_window_attn(const float* __restrict__ x, const bf16* __restrict__ ws,
                  const float* __restrict__ b_qkv1, const float* __restrict__ g_qkv,
                  const float* __restrict__ be_qkv, const float* __restrict__ b_qkv2,
                  const float* __restrict__ rpb,
                  const float* __restrict__ b_p1, const float* __restrict__ g_p,
                  const float* __restrict__ be_p, const float* __restrict__ b_p2,
                  float* __restrict__ out) {
    __shared__ __align__(16) unsigned char smem[SMEM_BYTES];
    bf16*  Pbuf  = (bf16*)(smem + P_OFF);
    bf16*  h1b   = (bf16*)(smem + 0);            // aliases Pbuf region; disjoint lifetimes
    bf16*  qkbuf = (bf16*)(smem + QK_OFF);
    bf16*  vTbuf = (bf16*)(smem + VT_OFF);
    bf16*  OUTb  = (bf16*)(smem + OUT_OFF);      // attn output; also staging buffer for x (xb)
    bf16*  xb    = OUTb;                         // alias: x staging lives before attention
    float* psum  = (float*)(smem + ST_OFF);
    float* psq   = psum + 256;
    float* meanb = psq + 256;
    float* rstdb = meanb + 64;

    const int tid  = threadIdx.x;
    const int wave = tid >> 5;
    const int lane = tid & 31;
    const int lh   = lane >> 4;     // 16-lane half
    const int ln   = lane & 15;
    const int win  = blockIdx.x;
    const float* xw = x + (size_t)win * NTOK * DIM;

    const bf16* wt1 = ws + WT1_OFF;
    const bf16* wt2 = ws + WT2_OFF;
    const bf16* wt3 = ws + WT3_OFF;
    const bf16* wt4 = ws + WT4_OFF;

    const int mt = wave >> 1;       // each pair of waves shares an m-tile row

    // ---------------- stage x (fp32 global) -> xb (bf16 LDS) ----------------
    {
#pragma clang loop unroll(disable)
        for (int e = tid * 4; e < NTOK * DIM; e += 256 * 4) {
            const float4 v = *(const float4*)(xw + e);
            const int row = e >> 8, col = e & 255;
            *(v4bf*)(xb + row * OUT_PITCH + col) =
                (v4bf){(bf16)v.x, (bf16)v.y, (bf16)v.z, (bf16)v.w};
        }
    }
    __syncthreads();

    // ---------------- GEMM1: h1 = x[64x256] @ W1[256x512] ----------------
    {
#pragma clang loop unroll(disable)
        for (int j = 0; j < 16; ++j) {
            const int n0 = ((wave & 1) * 16 + j) * 16;
            if (j + 1 < 16)
                __builtin_prefetch(wt1 + (size_t)(n0 + 16 + ln) * 256, 0, 0);
            v8f c = V8F_ZERO;
#pragma unroll 4
            for (int kt = 0; kt < 8; ++kt)
                c = wmma_bf16(a_frag_lds(xb, OUT_PITCH, mt * 16, kt * 32, lane),
                              b_frag_g(wt1, 256, n0, kt * 32, lane), c);
            const int ncol = n0 + ln;
            const float bias = b_qkv1[ncol];
#pragma unroll
            for (int r = 0; r < 8; ++r) {
                const int M = mt * 16 + r + 8 * lh;
                h1b[M * H1_PITCH + ncol] = (bf16)(c[r] + bias);
            }
        }
    }
    // ---------------- LN + GELU on h1 (512 cols) ----------------
    ln_gelu_inplace(h1b, H1_PITCH, 512, g_qkv, be_qkv, psum, psq, meanb, rstdb, tid);

    // ---------------- GEMM2: qkv = h1[64x512] @ W2[512x768] ----------------
    {
#pragma clang loop unroll(disable)
        for (int j = 0; j < 24; ++j) {
            const int n0 = ((wave & 1) * 24 + j) * 16;
            if (j + 1 < 24)
                __builtin_prefetch(wt2 + (size_t)(n0 + 16 + ln) * 512, 0, 0);
            v8f c = V8F_ZERO;
#pragma unroll 4
            for (int kt = 0; kt < 16; ++kt)
                c = wmma_bf16(a_frag_lds(h1b, H1_PITCH, mt * 16, kt * 32, lane),
                              b_frag_g(wt2, 512, n0, kt * 32, lane), c);
            const int ncol = n0 + ln;
            const float bias = b_qkv2[ncol];
            if (n0 < 256) {                         // q: row-major
#pragma unroll
                for (int r = 0; r < 8; ++r) {
                    const int M = mt * 16 + r + 8 * lh;
                    qkbuf[M * QK_PITCH + ncol] = (bf16)(c[r] + bias);
                }
            } else if (n0 < 512) {                  // k: row-major at col offset 264
#pragma unroll
                for (int r = 0; r < 8; ++r) {
                    const int M = mt * 16 + r + 8 * lh;
                    qkbuf[M * QK_PITCH + 264 + (ncol - 256)] = (bf16)(c[r] + bias);
                }
            } else {                                // v: transposed per head -> vT[h][d][row]
                const int d = ncol - 512, hh = d >> 5, dd = d & 31;
#pragma unroll
                for (int r = 0; r < 8; ++r) {
                    const int M = mt * 16 + r + 8 * lh;
                    vTbuf[(hh * 32 + dd) * P_PITCH + M] = (bf16)(c[r] + bias);
                }
            }
        }
    }
    __syncthreads();

    // ---------------- Attention: one head per wave ----------------
    {
        const int h = wave;
        const bf16* kbase = qkbuf + 264 + h * HD;
        bf16* Ph = Pbuf + h * NTOK * P_PITCH;
        // Q.K^T one q-tile row at a time (keeps only 4 accumulator tiles live)
#pragma clang loop unroll(disable)
        for (int qt = 0; qt < 4; ++qt) {
            v16bf a = a_frag_lds(qkbuf, QK_PITCH, qt * 16, h * HD, lane);   // K dim = 32
            v8f cr[4];
#pragma unroll
            for (int nt = 0; nt < 4; ++nt) {
                v8f z = V8F_ZERO;
                cr[nt] = wmma_bf16(a, b_frag_lds(kbase, QK_PITCH, nt * 16, 0, lane), z);
            }
            // softmax with rel-pos bias, register-resident (16-lane group reductions)
#pragma unroll
            for (int r = 0; r < 8; ++r) {
                const int M = qt * 16 + r + 8 * lh;
                const int mr = M >> 3, mc = M & 7;
                float vals[4];
#pragma unroll
                for (int nt = 0; nt < 4; ++nt) {
                    const int Nn = nt * 16 + ln;
                    const int dr = mr - (Nn >> 3) + 7, dc = mc - (Nn & 7) + 7;
                    vals[nt] = cr[nt][r] * SCALE + rpb[(dr * 15 + dc) * NH + h];
                }
                float rm = fmaxf(fmaxf(vals[0], vals[1]), fmaxf(vals[2], vals[3]));
#pragma unroll
                for (int msk = 1; msk < 16; msk <<= 1) rm = fmaxf(rm, __shfl_xor(rm, msk, 32));
                float rs = 0.f;
#pragma unroll
                for (int nt = 0; nt < 4; ++nt) { vals[nt] = __expf(vals[nt] - rm); rs += vals[nt]; }
#pragma unroll
                for (int msk = 1; msk < 16; msk <<= 1) rs += __shfl_xor(rs, msk, 32);
                const float inv = 1.0f / rs;
#pragma unroll
                for (int nt = 0; nt < 4; ++nt) Ph[M * P_PITCH + nt * 16 + ln] = (bf16)(vals[nt] * inv);
            }
        }
        // out_head = P[64x64] @ V[64x32]
        const bf16* vTh = vTbuf + h * HD * P_PITCH;
#pragma clang loop unroll(disable)
        for (int qt = 0; qt < 4; ++qt) {
#pragma unroll
            for (int nt = 0; nt < 2; ++nt) {
                v8f acc = V8F_ZERO;
#pragma unroll
                for (int kb = 0; kb < 2; ++kb)
                    acc = wmma_bf16(a_frag_lds(Ph, P_PITCH, qt * 16, kb * 32, lane),
                                    b_frag_lds(vTh, P_PITCH, nt * 16, kb * 32, lane), acc);
#pragma unroll
                for (int r = 0; r < 8; ++r) {
                    const int M = qt * 16 + r + 8 * lh;
                    OUTb[M * OUT_PITCH + h * HD + nt * 16 + ln] = (bf16)acc[r];
                }
            }
        }
    }
    __syncthreads();

    // ---------------- GEMM3: h2 = attn_out[64x256] @ W3[256x512] ----------------
    {
#pragma clang loop unroll(disable)
        for (int j = 0; j < 16; ++j) {
            const int n0 = ((wave & 1) * 16 + j) * 16;
            if (j + 1 < 16)
                __builtin_prefetch(wt3 + (size_t)(n0 + 16 + ln) * 256, 0, 0);
            v8f c = V8F_ZERO;
#pragma unroll 4
            for (int kt = 0; kt < 8; ++kt)
                c = wmma_bf16(a_frag_lds(OUTb, OUT_PITCH, mt * 16, kt * 32, lane),
                              b_frag_g(wt3, 256, n0, kt * 32, lane), c);
            const int ncol = n0 + ln;
            const float bias = b_p1[ncol];
#pragma unroll
            for (int r = 0; r < 8; ++r) {
                const int M = mt * 16 + r + 8 * lh;
                h1b[M * H1_PITCH + ncol] = (bf16)(c[r] + bias);   // h2 reuses h1 slot
            }
        }
    }
    // ---------------- LN + GELU on h2 (512 cols) ----------------
    ln_gelu_inplace(h1b, H1_PITCH, 512, g_p, be_p, psum, psq, meanb, rstdb, tid);

    // ---------------- GEMM4: y = h2[64x512] @ W4[512x256] -> global fp32 ----------------
    {
#pragma clang loop unroll(disable)
        for (int j = 0; j < 8; ++j) {
            const int n0 = ((wave & 1) * 8 + j) * 16;
            if (j + 1 < 8)
                __builtin_prefetch(wt4 + (size_t)(n0 + 16 + ln) * 512, 0, 0);
            v8f c = V8F_ZERO;
#pragma unroll 4
            for (int kt = 0; kt < 16; ++kt)
                c = wmma_bf16(a_frag_lds(h1b, H1_PITCH, mt * 16, kt * 32, lane),
                              b_frag_g(wt4, 512, n0, kt * 32, lane), c);
            const int ncol = n0 + ln;
            const float bias = b_p2[ncol];
#pragma unroll
            for (int r = 0; r < 8; ++r) {
                const int M = mt * 16 + r + 8 * lh;
                out[((size_t)win * NTOK + M) * DIM + ncol] = c[r] + bias;
            }
        }
    }
}

extern "C" void kernel_launch(void* const* d_in, const int* in_sizes, int n_in,
                              void* d_out, int out_size, void* d_ws, size_t ws_size,
                              hipStream_t stream) {
    (void)in_sizes; (void)n_in; (void)out_size; (void)ws_size;
    const float* x      = (const float*)d_in[0];
    const float* w_qkv1 = (const float*)d_in[1];
    const float* b_qkv1 = (const float*)d_in[2];
    const float* g_qkv  = (const float*)d_in[3];
    const float* be_qkv = (const float*)d_in[4];
    const float* w_qkv2 = (const float*)d_in[5];
    const float* b_qkv2 = (const float*)d_in[6];
    const float* rpb    = (const float*)d_in[7];
    const float* w_p1   = (const float*)d_in[8];
    const float* b_p1   = (const float*)d_in[9];
    const float* g_p    = (const float*)d_in[10];
    const float* be_p   = (const float*)d_in[11];
    const float* w_p2   = (const float*)d_in[12];
    const float* b_p2   = (const float*)d_in[13];
    float* out = (float*)d_out;
    bf16* wsb  = (bf16*)d_ws;

    prep_weights<<<(WT_TOTAL + 255) / 256, 256, 0, stream>>>(w_qkv1, w_qkv2, w_p1, w_p2, wsb);
    fused_window_attn<<<NWIN, 256, 0, stream>>>(x, wsb, b_qkv1, g_qkv, be_qkv, b_qkv2, rpb,
                                                b_p1, g_p, be_p, b_p2, out);
}
